// PNAGNN_73297911873709
// MI455X (gfx1250) — matrix operand order repaired and checked by reading, not compiled, not run on hardware
//
#include <hip/hip_runtime.h>
#include <stdint.h>

// ---------------------------------------------------------------------------
// PNA GNN layer for gfx1250 (MI455X), bf16 WMMA + fp32 accumulation.
// Pretrans weights staged to LDS via async global->LDS (ASYNCcnt) copies.
// ---------------------------------------------------------------------------

#define NNODES   50000
#define NEDGES   800000
#define DIM      128
#define PRE_K    384          // 2*DIM + EDGE_DIM
#define POST_K   1664         // 13*DIM
#define AVG_D_LOG 2.833213344056216f   // log(17.0)
#define EPSV     1e-5f

typedef __attribute__((ext_vector_type(16))) __bf16 v16bf;
typedef __attribute__((ext_vector_type(8)))  float  v8f;

union FragB16 { v16bf v; uint32_t u[8]; uint4 q[2]; };

__device__ __forceinline__ uint32_t bf16_1(float x){
    uint32_t a = __float_as_uint(x);
    return (a + 0x7FFFu + ((a >> 16) & 1u)) >> 16;   // RNE to bf16
}
__device__ __forceinline__ uint32_t bf16pair(float lo, float hi){
    return (bf16_1(lo) & 0xFFFFu) | (bf16_1(hi) << 16);
}
__device__ __forceinline__ v8f vzero(){
    v8f z;
#pragma unroll
    for (int i = 0; i < 8; ++i) z[i] = 0.0f;
    return z;
}
__device__ __forceinline__ v8f bwmma(const FragB16& A, const FragB16& B, v8f c){
    return __builtin_amdgcn_wmma_f32_16x16x32_bf16(false, A.v, false, B.v,
                                                   (short)0, c, false, false);
}

// ---------------------------------------------------------------------------
// Prep kernels
// ---------------------------------------------------------------------------
__global__ void k_init(float* asum, float* asq, uint32_t* amax, uint32_t* amin,
                       float* deg)
{
    int t = blockIdx.x * blockDim.x + threadIdx.x;
    if (t < NNODES * DIM) {
        asum[t] = 0.0f;
        asq[t]  = 0.0f;
        amax[t] = 0u;            // +0.0f (messages are post-ReLU, >= 0)
        amin[t] = 0x7F800000u;   // +inf
    }
    if (t < NNODES) deg[t] = 0.0f;
}

__global__ void k_cvt_pairs(const float* __restrict__ in, uint32_t* __restrict__ out,
                            int npairs)
{
    int t = blockIdx.x * blockDim.x + threadIdx.x;
    if (t < npairs) out[t] = bf16pair(in[2 * t], in[2 * t + 1]);
}

// Pack weight matrix W[K,128] (row major) into per-lane B-fragment order:
// out[((kt*8 + nt)*32 + lane)*8 + v] holds bf16{W[k][n], W[k+1][n]},
//   n = nt*16 + lane%16,  k = kt*32 + 8*(lane/16) + 2*v + (v >= 4 ? 8 : 0).
__global__ void k_pack_w(const float* __restrict__ W, uint32_t* __restrict__ out,
                         int ndwords)
{
    int idx = blockIdx.x * blockDim.x + threadIdx.x;
    if (idx >= ndwords) return;
    int kt   = idx >> 11;
    int rem  = idx & 2047;
    int nt   = rem >> 8;
    int rem2 = rem & 255;
    int lane = rem2 >> 3;
    int v    = rem2 & 7;
    int n = nt * 16 + (lane & 15);
    int k = kt * 32 + (lane >> 4) * 8 + 2 * v + (v >= 4 ? 8 : 0);
    out[idx] = bf16pair(W[k * DIM + n], W[(k + 1) * DIM + n]);
}

__global__ void k_degree(const int* __restrict__ dst, float* __restrict__ deg)
{
    int t = blockIdx.x * blockDim.x + threadIdx.x;
    if (t < NEDGES) atomicAdd(&deg[dst[t]], 1.0f);
}

// ---------------------------------------------------------------------------
// Pretrans: msg = relu([nf[src], nf[dst], ef] @ w_pre + b_pre), fused with
// multi-aggregator scatter (sum, sum^2, max, min) via atomics.
// One wave = 16 edges x 128 out dims; 8 waves/block -> 128 edges/block.
// Packed w_pre (96KB) staged to LDS once per block via async global->LDS.
// ---------------------------------------------------------------------------
__global__ __launch_bounds__(256)
void k_pretrans(const uint32_t* __restrict__ nf16,
                const float*    __restrict__ edge_feat,
                const uint32_t* __restrict__ wpk,
                const float*    __restrict__ b_pre,
                const int*      __restrict__ src,
                const int*      __restrict__ dst,
                float* __restrict__ asum, float* __restrict__ asq,
                uint32_t* __restrict__ amax, uint32_t* __restrict__ amin)
{
    extern __shared__ uint32_t wlds[];   // PRE_K*64 dwords = 96KB (dynamic)

    // --- stage packed w_pre into LDS: async copies, ASYNCcnt tracked ---
    {
        // flat LDS addresses: low 32 bits = allocation-relative byte offset
        uint32_t lbase = (uint32_t)(uintptr_t)(void*)wlds;
        const char* gbase = (const char*)wpk;
#pragma unroll
        for (int i = 0; i < 24; ++i) {               // 256 thr * 24 * 16B = 96KB
            int idx = threadIdx.x + i * 256;         // uint4 index
            uint32_t la = lbase + (uint32_t)idx * 16u;
            uint64_t ga = (uint64_t)(uintptr_t)(gbase + (size_t)idx * 16);
            asm volatile("global_load_async_to_lds_b128 %0, %1, off"
                         :: "v"(la), "v"(ga) : "memory");
        }
        asm volatile("s_wait_asynccnt 0x0" ::: "memory");
    }
    __syncthreads();

    int lane = threadIdx.x & 31;
    int wv   = threadIdx.x >> 5;
    int base = (blockIdx.x * 8 + wv) * 16;
    int half = lane >> 4;
    int m    = lane & 15;
    int e    = base + m;

    int sN = src[e];
    int dN = dst[e];
    const uint32_t* srow = nf16 + sN * (DIM / 2);
    const uint32_t* drow = nf16 + dN * (DIM / 2);
    const float*    erow = edge_feat + e * DIM;

    v8f acc[8];
#pragma unroll
    for (int i = 0; i < 8; ++i) acc[i] = vzero();

#pragma unroll
    for (int kt = 0; kt < 12; ++kt) {
        FragB16 A;
        int seg  = kt >> 2;                       // 0:src 1:dst 2:edge
        int off1 = (kt & 3) * 32 + half * 8;      // first 8-halfword run
        int off2 = off1 + 16;                     // second run
        if (seg == 0) {
            A.q[0] = *(const uint4*)(srow + (off1 >> 1));
            A.q[1] = *(const uint4*)(srow + (off2 >> 1));
        } else if (seg == 1) {
            A.q[0] = *(const uint4*)(drow + (off1 >> 1));
            A.q[1] = *(const uint4*)(drow + (off2 >> 1));
        } else {
            float4 f0 = *(const float4*)(erow + off1);
            float4 f1 = *(const float4*)(erow + off1 + 4);
            float4 f2 = *(const float4*)(erow + off2);
            float4 f3 = *(const float4*)(erow + off2 + 4);
            A.u[0] = bf16pair(f0.x, f0.y); A.u[1] = bf16pair(f0.z, f0.w);
            A.u[2] = bf16pair(f1.x, f1.y); A.u[3] = bf16pair(f1.z, f1.w);
            A.u[4] = bf16pair(f2.x, f2.y); A.u[5] = bf16pair(f2.z, f2.w);
            A.u[6] = bf16pair(f3.x, f3.y); A.u[7] = bf16pair(f3.z, f3.w);
        }
#pragma unroll
        for (int nt = 0; nt < 8; ++nt) {
            FragB16 B;
            const uint32_t* wp = wlds + ((kt * 8 + nt) * 32 + lane) * 8;
            B.q[0] = *(const uint4*)(wp);
            B.q[1] = *(const uint4*)(wp + 4);
            acc[nt] = bwmma(A, B, acc[nt]);
        }
    }

    // epilogue: bias + relu + scatter-reduce per destination node
    int dr[8];
#pragma unroll
    for (int r = 0; r < 8; ++r) dr[r] = dst[base + r + 8 * half];

#pragma unroll
    for (int nt = 0; nt < 8; ++nt) {
        int n = nt * 16 + (lane & 15);
        float bias = b_pre[n];
#pragma unroll
        for (int r = 0; r < 8; ++r) {
            float val = acc[nt][r] + bias;
            val = fmaxf(val, 0.0f);
            int idx = dr[r] * DIM + n;
            if (val > 0.0f) {                     // 0 is identity for add/max
                atomicAdd(&asum[idx], val);
                atomicAdd(&asq[idx],  val * val);
                atomicMax(&amax[idx], __float_as_uint(val));
            }
            atomicMin(&amin[idx], __float_as_uint(val)); // vals >= 0: bits monotone
        }
    }
}

// ---------------------------------------------------------------------------
// Finalize aggregators: mean/max/min/std -> bf16 agg table; degree scalers.
// ---------------------------------------------------------------------------
__global__ void k_finalize(const float* __restrict__ asum, const float* __restrict__ asq,
                           const uint32_t* __restrict__ amax, const uint32_t* __restrict__ amin,
                           const float* __restrict__ deg,
                           uint32_t* __restrict__ agg16,
                           float* __restrict__ s_a, float* __restrict__ s_t)
{
    int t = blockIdx.x * blockDim.x + threadIdx.x;
    if (t >= NNODES * 64) return;
    int node = t >> 6;
    int p    = t & 63;
    int j    = p * 2;

    float dg = deg[node];
    float ds = fmaxf(dg, 1.0f);
    bool  nb = dg > 0.0f;
    int idx = node * DIM + j;

    float m0 = asum[idx] / ds,     m1 = asum[idx + 1] / ds;
    float v0 = fmaxf(asq[idx] / ds - m0 * m0, 0.0f);
    float v1 = fmaxf(asq[idx + 1] / ds - m1 * m1, 0.0f);
    float sd0 = sqrtf(v0 + EPSV),  sd1 = sqrtf(v1 + EPSV);
    float mx0 = nb ? __uint_as_float(amax[idx])     : 0.0f;
    float mx1 = nb ? __uint_as_float(amax[idx + 1]) : 0.0f;
    float mn0 = nb ? __uint_as_float(amin[idx])     : 0.0f;
    float mn1 = nb ? __uint_as_float(amin[idx + 1]) : 0.0f;

    uint32_t* ag = agg16 + node * 256;            // 512 bf16 = 256 dwords/node
    ag[p]        = bf16pair(m0,  m1);             // mean
    ag[64 + p]   = bf16pair(mx0, mx1);            // max
    ag[128 + p]  = bf16pair(mn0, mn1);            // min
    ag[192 + p]  = bf16pair(sd0, sd1);            // std

    if (p == 0) {
        float logd = logf(dg + 1.0f);
        s_a[node] = logd / AVG_D_LOG;
        s_t[node] = AVG_D_LOG / fmaxf(logd, EPSV);
    }
}

// ---------------------------------------------------------------------------
// Posttrans: h1 = relu(nf@Wnf + agg@Wagg + s_a*(agg@Wamp) + s_t*(agg@Watt) + b1)
//            out = h1 @ w_post2 + b2 + nf   (second GEMM via LDS re-fragment)
// One wave = 16 nodes; 8 waves/block.
// ---------------------------------------------------------------------------
__global__ __launch_bounds__(256)
void k_posttrans(const uint32_t* __restrict__ nf16,
                 const uint32_t* __restrict__ agg16,
                 const float* __restrict__ s_a, const float* __restrict__ s_t,
                 const uint32_t* __restrict__ wp1, const uint32_t* __restrict__ wp2,
                 const float* __restrict__ b1,  const float* __restrict__ b2,
                 const float* __restrict__ node_feat, float* __restrict__ out)
{
    __shared__ __align__(16) unsigned short h1s[8][16][DIM];

    int lane = threadIdx.x & 31;
    int wv   = threadIdx.x >> 5;
    int tile = blockIdx.x * 8 + wv;
    if (tile >= NNODES / 16) return;              // no block barriers used
    int base = tile * 16;
    int half = lane >> 4;
    int m    = lane & 15;

    const uint32_t* nfrow = nf16  + (size_t)(base + m) * (DIM / 2);
    const uint32_t* agrow = agg16 + (size_t)(base + m) * 256;

    float sa[8], st[8];
#pragma unroll
    for (int r = 0; r < 8; ++r) {
        int nd = base + r + 8 * half;
        sa[r] = s_a[nd];
        st[r] = s_t[nd];
    }

    for (int nt = 0; nt < 8; ++nt) {
        v8f aP = vzero(), aA = vzero(), aB = vzero(), aC = vzero();

        // node_feat segment: K = 128 (weight kt 0..3)
#pragma unroll
        for (int s = 0; s < 4; ++s) {
            FragB16 A;
            int off1 = s * 32 + half * 8, off2 = off1 + 16;
            A.q[0] = *(const uint4*)(nfrow + (off1 >> 1));
            A.q[1] = *(const uint4*)(nfrow + (off2 >> 1));
            FragB16 B;
            const uint32_t* wp = wp1 + (size_t)((s * 8 + nt) * 32 + lane) * 8;
            B.q[0] = *(const uint4*)(wp);
            B.q[1] = *(const uint4*)(wp + 4);
            aP = bwmma(A, B, aP);
        }
        // agg segment: K = 512 shared across agg/amp/att weight blocks
#pragma unroll
        for (int s = 0; s < 16; ++s) {
            FragB16 A;
            int off1 = s * 32 + half * 8, off2 = off1 + 16;
            A.q[0] = *(const uint4*)(agrow + (off1 >> 1));
            A.q[1] = *(const uint4*)(agrow + (off2 >> 1));
            FragB16 B;
            const uint32_t* wpA = wp1 + (size_t)(((4 + s) * 8 + nt) * 32 + lane) * 8;
            B.q[0] = *(const uint4*)(wpA); B.q[1] = *(const uint4*)(wpA + 4);
            aA = bwmma(A, B, aA);
            const uint32_t* wpB = wp1 + (size_t)(((20 + s) * 8 + nt) * 32 + lane) * 8;
            B.q[0] = *(const uint4*)(wpB); B.q[1] = *(const uint4*)(wpB + 4);
            aB = bwmma(A, B, aB);
            const uint32_t* wpC = wp1 + (size_t)(((36 + s) * 8 + nt) * 32 + lane) * 8;
            B.q[0] = *(const uint4*)(wpC); B.q[1] = *(const uint4*)(wpC + 4);
            aC = bwmma(A, B, aC);
        }

        int n = nt * 16 + (lane & 15);
        float bias = b1[n];
#pragma unroll
        for (int r = 0; r < 8; ++r) {
            float h = aP[r] + aA[r] + sa[r] * aB[r] + st[r] * aC[r] + bias;
            h = fmaxf(h, 0.0f);
            h1s[wv][r + 8 * half][n] = (unsigned short)bf16_1(h);
        }
    }

    // second GEMM: h1[16x128] @ w_post2[128x128], wave-private LDS (DS ops
    // from the same wave are in-order; backend inserts s_wait_dscnt).
#pragma unroll
    for (int nt = 0; nt < 8; ++nt) {
        v8f a2 = vzero();
#pragma unroll
        for (int s = 0; s < 4; ++s) {
            FragB16 A;
            int off1 = s * 32 + half * 8, off2 = off1 + 16;
            A.q[0] = *(const uint4*)&h1s[wv][m][off1];
            A.q[1] = *(const uint4*)&h1s[wv][m][off2];
            FragB16 B;
            const uint32_t* wp = wp2 + (size_t)((s * 8 + nt) * 32 + lane) * 8;
            B.q[0] = *(const uint4*)(wp);
            B.q[1] = *(const uint4*)(wp + 4);
            a2 = bwmma(A, B, a2);
        }
        int n = nt * 16 + (lane & 15);
        float bias = b2[n];
#pragma unroll
        for (int r = 0; r < 8; ++r) {
            int nd = base + r + 8 * half;
            out[nd * DIM + n] = a2[r] + bias + node_feat[nd * DIM + n];
        }
    }
}

// ---------------------------------------------------------------------------
// Launch
// ---------------------------------------------------------------------------
extern "C" void kernel_launch(void* const* d_in, const int* in_sizes, int n_in,
                              void* d_out, int out_size, void* d_ws, size_t ws_size,
                              hipStream_t stream)
{
    (void)in_sizes; (void)n_in; (void)out_size; (void)ws_size;

    const float* node_feat = (const float*)d_in[0];
    const float* edge_feat = (const float*)d_in[1];
    const float* w_pre     = (const float*)d_in[2];
    const float* b_pre     = (const float*)d_in[3];
    const float* w_post1   = (const float*)d_in[4];
    const float* b_post1   = (const float*)d_in[5];
    const float* w_post2   = (const float*)d_in[6];
    const float* b_post2   = (const float*)d_in[7];
    const int*   src       = (const int*)d_in[8];
    const int*   dst       = (const int*)d_in[9];
    float*       out       = (float*)d_out;

    char* ws = (char*)d_ws;
    size_t off = 0;
    auto carve = [&](size_t bytes) -> void* {
        void* p = ws + off;
        off += (bytes + 255) & ~(size_t)255;
        return p;
    };
    float*    asum    = (float*)   carve((size_t)NNODES * DIM * 4);
    float*    asq     = (float*)   carve((size_t)NNODES * DIM * 4);
    uint32_t* amax    = (uint32_t*)carve((size_t)NNODES * DIM * 4);
    uint32_t* amin    = (uint32_t*)carve((size_t)NNODES * DIM * 4);
    float*    deg     = (float*)   carve((size_t)NNODES * 4);
    float*    sA      = (float*)   carve((size_t)NNODES * 4);
    float*    sT      = (float*)   carve((size_t)NNODES * 4);
    uint32_t* nf16    = (uint32_t*)carve((size_t)NNODES * DIM * 2);
    uint32_t* agg16   = (uint32_t*)carve((size_t)NNODES * 512 * 2);
    uint32_t* wpre_pk = (uint32_t*)carve((size_t)PRE_K  * 64 * 4);
    uint32_t* wp1_pk  = (uint32_t*)carve((size_t)POST_K * 64 * 4);
    uint32_t* wp2_pk  = (uint32_t*)carve((size_t)DIM    * 64 * 4);

    // prep
    k_init<<<(NNODES * DIM + 255) / 256, 256, 0, stream>>>(asum, asq, amax, amin, deg);
    k_cvt_pairs<<<(NNODES * 64 + 255) / 256, 256, 0, stream>>>(node_feat, nf16, NNODES * 64);
    k_pack_w<<<(PRE_K * 64 + 255) / 256, 256, 0, stream>>>(w_pre,   wpre_pk, PRE_K * 64);
    k_pack_w<<<(POST_K * 64 + 255) / 256, 256, 0, stream>>>(w_post1, wp1_pk, POST_K * 64);
    k_pack_w<<<(DIM * 64 + 255) / 256, 256, 0, stream>>>(w_post2, wp2_pk, DIM * 64);
    k_degree<<<(NEDGES + 255) / 256, 256, 0, stream>>>(dst, deg);

    // edge GEMM + fused scatter-reduce (6250 blocks * 128 edges = 800000),
    // 96KB dynamic LDS for the async-staged packed w_pre.
    k_pretrans<<<NEDGES / 128, 256, PRE_K * 64 * 4, stream>>>(
        nf16, edge_feat, wpre_pk, b_pre, src, dst, asum, asq, amax, amin);

    // aggregator finalize
    k_finalize<<<(NNODES * 64 + 255) / 256, 256, 0, stream>>>(asum, asq, amax, amin,
                                                              deg, agg16, sA, sT);

    // node GEMMs + residual (3125 node tiles, 8 per block)
    k_posttrans<<<(NNODES / 16 + 7) / 8, 256, 0, stream>>>(nf16, agg16, sA, sT,
                                                           wp1_pk, wp2_pk,
                                                           b_post1, b_post2,
                                                           node_feat, out);
}
